// NewtonNet_56367150793521
// MI455X (gfx1250) — matrix operand order repaired and checked by reading, not compiled, not run on hardware
//
#include <hip/hip_runtime.h>
#include <math.h>

typedef __attribute__((ext_vector_type(16))) _Float16 v16h;
typedef __attribute__((ext_vector_type(8)))  _Float16 v8h;
typedef __attribute__((ext_vector_type(4)))  _Float16 v4h;
typedef __attribute__((ext_vector_type(8)))  float    v8f;
typedef __attribute__((ext_vector_type(4)))  float    v4f;

#define NB_B 4
#define NB_A 512
#define NB_M 64
#define NB_F 128
#define CUTF 5.0f
#define PIF  3.14159265358979323846f

// LDS strides (halfs): 16-byte aligned rows for ds_load_b128 fragment loads
#define LDA_T 136   // 64-row activation tiles: 272 B/row = 17*16
#define LDW_T 72    // transposed weight tile  : 144 B/col = 9*16
#define LDR_T 40    // rbf tile                : 80 B/row  = 5*16

// packed f16 weight offsets (halfs) inside the per-layer weight arena
#define WOFF_IMN1 0
#define WOFF_IMN2 16384
#define WOFF_ESU1 32768
#define WOFF_ESU2 49152
#define WOFF_ISU1 65536
#define WOFF_ISU2 81920
#define WOFF_EMF1 98304
#define WOFF_EMF2 114688
#define WOFF_EME1 131072
#define WOFF_EME2 147456
#define WOFF_IME  163840   // 128*64 padded (K 20 -> 64)
#define WARENA_H  172032

__device__ __forceinline__ float silu_f(float x) { return x / (1.0f + expf(-x)); }

// ---------- WMMA fragment loaders (CDNA5 16x16x32 f16 layouts) ----------
// A (16x32): lane&15 = M row; two contiguous 16B LDS loads per lane.
__device__ __forceinline__ v16h load_frag_a(const _Float16* Ah, int lda, int r0, int kb, int lane) {
  int row = r0 + (lane & 15);
  int khi = (lane >> 4) & 1;
  const _Float16* p = Ah + row * lda + kb + khi * 8;
  union { v16h v; struct { v8h lo; v8h hi; } s; } u;
  u.s.lo = *(const v8h*)p;
  u.s.hi = *(const v8h*)(p + 16);
  return u.v;
}
// B (32x16) from LDS tile stored TRANSPOSED as Wt[col][k]: two 16B loads.
__device__ __forceinline__ v16h load_frag_bt(const _Float16* Wt, int kb, int c0, int lane) {
  int col = c0 + (lane & 15);
  int khi = (lane >> 4) & 1;
  const _Float16* p = Wt + col * LDW_T + kb + khi * 8;
  union { v16h v; struct { v8h lo; v8h hi; } s; } u;
  u.s.lo = *(const v8h*)p;
  u.s.hi = *(const v8h*)(p + 16);
  return u.v;
}

// 64xNCOL = A[64xK](LDS f16, lda) @ W (pre-packed f16, chunked-transposed
// [k0/64][n][k&63]); stage per 64-K chunk is a pure b128 copy into `wgt`.
template<int NCOL>
__device__ __forceinline__ void gemm_tiles(const _Float16* Ah, int lda,
                                           const _Float16* __restrict__ Wt_g, int K,
                                           _Float16* wgt, v8f* acc,
                                           int t, int w, int lane) {
  constexpr int TJ = NCOL / 32;
  int r0 = (w >> 1) * 16;
  int cbase = (w & 1) * (NCOL / 2);
#pragma unroll
  for (int j = 0; j < TJ; ++j)
#pragma unroll
    for (int r = 0; r < 8; ++r) acc[j][r] = 0.0f;
  for (int k0 = 0; k0 < K; k0 += 64) {
    __syncthreads();                       // prior users of wgt are done
    const _Float16* src = Wt_g + (long)(k0 >> 6) * (NCOL * 64);
    for (int idx = t; idx < NCOL * 8; idx += 256) {
      int n = idx >> 3, r8 = (idx & 7) << 3;
      *(v8h*)&wgt[n * LDW_T + r8] = *(const v8h*)&src[n * 64 + r8];
    }
    __syncthreads();
#pragma unroll
    for (int kb = 0; kb < 64; kb += 32) {
      v16h a = load_frag_a(Ah, lda, r0, k0 + kb, lane);
#pragma unroll
      for (int j = 0; j < TJ; ++j) {
        v16h bf = load_frag_bt(wgt, kb, cbase + j * 16, lane);
        acc[j] = __builtin_amdgcn_wmma_f32_16x16x32_f16(
            false, a, false, bf, (short)0, acc[j], false, false);
      }
    }
  }
}

// stage 64 rows of f32 [.,128] global into f16 LDS tile (vectorized 4-wide)
__device__ __forceinline__ void stage_rows_f16(const float* __restrict__ X, long row0,
                                               _Float16* Xh, int t) {
  for (int idx = t; idx < 64 * 32; idx += 256) {
    int r = idx >> 5, c4 = (idx & 31) << 2;
    const float* xp = &X[(row0 + r) * 128 + c4];
    v4h h4;
    h4[0] = (_Float16)xp[0]; h4[1] = (_Float16)xp[1];
    h4[2] = (_Float16)xp[2]; h4[3] = (_Float16)xp[3];
    *(v4h*)&Xh[r * LDA_T + c4] = h4;
  }
}

// ---------- pack f32 weight [K,N] -> f16 chunked-transposed [ceil(K/64)][N][64] ----------
__global__ __launch_bounds__(256) void nn_pack_kernel(const float* __restrict__ W,
    _Float16* __restrict__ dst, int Ksrc, int Kpad, int N) {
  long idx = (long)blockIdx.x * 256 + threadIdx.x;
  if (idx < (long)Kpad * N) {
    int r = (int)(idx / N), n = (int)(idx % N);
    float v = (r < Ksrc) ? W[(long)r * N + n] : 0.0f;
    dst[((long)(r >> 6) * N + n) * 64 + (r & 63)] = (_Float16)v;
  }
}

// ---------- init: s = embed[z], zero f_eq/drA/out ----------
__global__ __launch_bounds__(256) void nn_init_kernel(const int* __restrict__ an,
    const float* __restrict__ embed, float* __restrict__ s,
    float* __restrict__ feq, float* __restrict__ drA, float* __restrict__ out) {
  long idx = (long)blockIdx.x * 256 + threadIdx.x;
  if (idx < (long)NB_B * NB_A * NB_F) {
    int atom = (int)(idx >> 7), f = (int)(idx & 127);
    s[idx] = embed[an[atom] * NB_F + f];
  }
  if (idx < (long)NB_B * NB_A * 3 * NB_F) { feq[idx] = 0.0f; drA[idx] = 0.0f; }
  if (idx < NB_B) out[idx] = 0.0f;
}

// ---------- Y = silu(X@W1+B1)@W2+B2 over 64 rows per workgroup ----------
__global__ __launch_bounds__(256) void nn_mlp2_kernel(const float* __restrict__ X,
    const _Float16* __restrict__ W1t, const float* __restrict__ B1,
    const _Float16* __restrict__ W2t, const float* __restrict__ B2,
    float* __restrict__ Y) {
  __shared__ __align__(16) _Float16 Xh[64 * LDA_T];
  __shared__ __align__(16) _Float16 wgt[128 * LDW_T];
  __shared__ __align__(16) _Float16 hid[64 * LDA_T];
  int t = threadIdx.x, w = t >> 5, lane = t & 31;
  int r0 = (w >> 1) * 16, cbase = (w & 1) * 64;
  int khi = (lane >> 4) & 1, cl = lane & 15;
  long row0 = (long)blockIdx.x * 64;
  stage_rows_f16(X, row0, Xh, t);
  v8f acc[4];
  gemm_tiles<128>(Xh, LDA_T, W1t, 128, wgt, acc, t, w, lane);
#pragma unroll
  for (int j = 0; j < 4; ++j) {
    int col = cbase + j * 16 + cl;
    float bb = B1[col];
#pragma unroll
    for (int r = 0; r < 8; ++r)
      hid[(r0 + r + khi * 8) * LDA_T + col] = (_Float16)silu_f(acc[j][r] + bb);
  }
  gemm_tiles<128>(hid, LDA_T, W2t, 128, wgt, acc, t, w, lane);
#pragma unroll
  for (int j = 0; j < 4; ++j) {
    int col = cbase + j * 16 + cl;
    float bb = B2[col];
#pragma unroll
    for (int r = 0; r < 8; ++r)
      Y[(row0 + r0 + r + khi * 8) * 128 + col] = acc[j][r] + bb;
  }
}

// ---------- per-atom megakernel ----------
__global__ __launch_bounds__(256) void nn_mega_kernel(
    const float* __restrict__ dist_g, const float* __restrict__ dvec_g,
    const float* __restrict__ mask_g, const int* __restrict__ nbr_g,
    const float* __restrict__ imn_g, float* __restrict__ s_g,
    float* __restrict__ feq_g, const float* __restrict__ drA_g,
    float* __restrict__ drB_g, float* __restrict__ updf_g,
    const _Float16* __restrict__ wime16, const float* __restrict__ bime_g,
    const float* __restrict__ wemc_g,
    const _Float16* __restrict__ emfw1_16, const float* __restrict__ emfb1_g,
    const _Float16* __restrict__ emfw2_16, const float* __restrict__ emfb2_g,
    const _Float16* __restrict__ emew1_16, const _Float16* __restrict__ emew2_16) {
  __shared__ __align__(16) _Float16 wgt[128 * LDW_T];
  __shared__ __align__(16) _Float16 msg[64 * LDA_T];
  __shared__ __align__(16) _Float16 hid[64 * LDA_T];
  __shared__ __align__(16) _Float16 rbf[64 * LDR_T];
  __shared__ __align__(16) float s_acc[128];
  __shared__ __align__(16) float imn_self[128];
  __shared__ float cutl[64];
  __shared__ float maskl[64];
  __shared__ float dhat[64 * 3];
  __shared__ float emf_m[64 * 3];   // (msg . w_emc) * dhat * mask
  __shared__ float updf_s[3 * 128];
  __shared__ int   nbl[64];

  int t = threadIdx.x, w = t >> 5, lane = t & 31;
  int r0 = (w >> 1) * 16, cbase = (w & 1) * 64;
  int khi = (lane >> 4) & 1, cl = lane & 15;
  int atom = blockIdx.x;
  int b = atom >> 9;                      // A = 512
  long ebase = (long)atom * NB_M;
  long bbaseF = (long)b * NB_A * NB_F;
  long bbase3 = (long)b * NB_A * 3 * NB_F;

  // ---- phase 0: edge data, RBF tile (K padded 20->32), w_ime stage (b128 copy)
  if (t < 64) {
    int m = t;
    float d = dist_g[ebase + m];
    maskl[m] = mask_g[ebase + m];
    cutl[m] = 0.5f * (cosf(PIF * d / CUTF) + 1.0f) * (d < CUTF ? 1.0f : 0.0f);
    float inv = 1.0f / (d + 1e-8f);
#pragma unroll
    for (int k = 0; k < 3; ++k) dhat[m * 3 + k] = dvec_g[(ebase + m) * 3 + k] * inv;
    nbl[m] = nbr_g[ebase + m];
  }
  if (t < 128) {
    imn_self[t] = imn_g[(long)atom * NB_F + t];
    s_acc[t] = s_g[(long)atom * NB_F + t];
  }
  for (int idx = t; idx < 64 * 32; idx += 256) {
    int m = idx >> 5, k = idx & 31;
    float v = 0.0f;
    if (k < 20) {
      float d = dist_g[ebase + m];
      v = sqrtf(2.0f / CUTF) * sinf((float)(k + 1) * PIF * d / CUTF) / d;
    }
    rbf[m * LDR_T + k] = (_Float16)v;
  }
  for (int idx = t; idx < 512; idx += 256) {  // first 32 K-rows of packed w_ime
    int n = idx >> 2, r8 = (idx & 3) << 3;
    *(v8h*)&wgt[n * LDW_T + r8] = *(const v8h*)&wime16[n * 64 + r8];
  }
  for (int idx = t; idx < 384; idx += 256) updf_s[idx] = 0.0f;
  __syncthreads();

  // ---- phase 1a: ime = (rbf@w_ime + b)*cut  -> hid (f16)
  {
    v8f acc[4];
#pragma unroll
    for (int j = 0; j < 4; ++j)
#pragma unroll
      for (int r = 0; r < 8; ++r) acc[j][r] = 0.0f;
    v16h a = load_frag_a(rbf, LDR_T, r0, 0, lane);
#pragma unroll
    for (int j = 0; j < 4; ++j) {
      v16h bf = load_frag_bt(wgt, 0, cbase + j * 16, lane);
      acc[j] = __builtin_amdgcn_wmma_f32_16x16x32_f16(false, a, false, bf, (short)0, acc[j], false, false);
    }
#pragma unroll
    for (int j = 0; j < 4; ++j) {
      int col = cbase + j * 16 + cl;
      float bb = bime_g[col];
#pragma unroll
      for (int r = 0; r < 8; ++r) {
        int row = r0 + r + khi * 8;
        hid[row * LDA_T + col] = (_Float16)((acc[j][r] + bb) * cutl[row]);
      }
    }
  }
  __syncthreads();

  // ---- phase 1b: msg = ime*imn_self*imn_nb ; s_acc += msg*mask
  // whole wave shares one edge m -> neighbor base goes to SGPR (saddr gather)
  for (int idx = t; idx < 64 * 32; idx += 256) {
    int m = idx >> 5, c4 = (idx & 31) << 2;
    int nbm = __builtin_amdgcn_readfirstlane(nbl[m]);
    float msk = maskl[m];
    const float* inb = imn_g + bbaseF + (long)nbm * NB_F + c4;
    v4f iv_nb = *(const v4f*)inb;
    v4f iv_self = *(const v4f*)&imn_self[c4];
    v4h ime4 = *(const v4h*)&hid[m * LDA_T + c4];
    v4h outp;
#pragma unroll
    for (int q = 0; q < 4; ++q) {
      float mv = (float)ime4[q] * iv_self[q] * iv_nb[q];
      outp[q] = (_Float16)mv;
      atomicAdd(&s_acc[c4 + q], mv * msk);
    }
    *(v4h*)&msg[m * LDA_T + c4] = outp;
  }
  __syncthreads();

  // ---- phase 2: write s_new; emF (masked) per edge
  if (t < 128) s_g[(long)atom * NB_F + t] = s_acc[t];
  if (t < 64) {
    float dot = 0.0f;
    const v8h* mrow = (const v8h*)&msg[t * LDA_T];
    for (int c8 = 0; c8 < 16; ++c8) {
      v8h mv = mrow[c8];
#pragma unroll
      for (int i = 0; i < 8; ++i) dot += (float)mv[i] * wemc_g[c8 * 8 + i];
    }
    float mm = dot * maskl[t];
#pragma unroll
    for (int k = 0; k < 3; ++k) emf_m[t * 3 + k] = mm * dhat[t * 3 + k];
  }
  __syncthreads();

  // ---- emf = MLP2(msg); fold upd_f out of the accumulators
  v8f acc[4];
  gemm_tiles<128>(msg, LDA_T, emfw1_16, 128, wgt, acc, t, w, lane);
#pragma unroll
  for (int j = 0; j < 4; ++j) {
    int col = cbase + j * 16 + cl;
    float bb = emfb1_g[col];
#pragma unroll
    for (int r = 0; r < 8; ++r)
      hid[(r0 + r + khi * 8) * LDA_T + col] = (_Float16)silu_f(acc[j][r] + bb);
  }
  gemm_tiles<128>(hid, LDA_T, emfw2_16, 128, wgt, acc, t, w, lane);
#pragma unroll
  for (int j = 0; j < 4; ++j) {
    int col = cbase + j * 16 + cl;
    float bb = emfb2_g[col];
    float p0 = 0.0f, p1 = 0.0f, p2 = 0.0f;
#pragma unroll
    for (int r = 0; r < 8; ++r) {
      int row = r0 + r + khi * 8;
      float e = acc[j][r] + bb;
      p0 += e * emf_m[row * 3 + 0];
      p1 += e * emf_m[row * 3 + 1];
      p2 += e * emf_m[row * 3 + 2];
    }
    atomicAdd(&updf_s[0 * 128 + col], p0);
    atomicAdd(&updf_s[1 * 128 + col], p1);
    atomicAdd(&updf_s[2 * 128 + col], p2);
  }
  __syncthreads();
  for (int idx = t; idx < 384; idx += 256) {
    float u = updf_s[idx];
    updf_g[(long)atom * 384 + idx] = u;
    feq_g[(long)atom * 384 + idx] += u;
  }

  // ---- eme = silu(msg@W1)@W2 (no biases); result overwrites msg (f16)
  gemm_tiles<128>(msg, LDA_T, emew1_16, 128, wgt, acc, t, w, lane);
#pragma unroll
  for (int j = 0; j < 4; ++j) {
    int col = cbase + j * 16 + cl;
#pragma unroll
    for (int r = 0; r < 8; ++r)
      hid[(r0 + r + khi * 8) * LDA_T + col] = (_Float16)silu_f(acc[j][r]);
  }
  gemm_tiles<128>(hid, LDA_T, emew2_16, 128, wgt, acc, t, w, lane);
#pragma unroll
  for (int j = 0; j < 4; ++j) {
    int col = cbase + j * 16 + cl;
#pragma unroll
    for (int r = 0; r < 8; ++r)
      msg[(r0 + r + khi * 8) * LDA_T + col] = (_Float16)acc[j][r];
  }
  __syncthreads();

  // ---- dr_tmp = dr_old + sum_m mask*eme[m]*dr_old[nb[m]]
  // thread f keeps 3 accumulators; m uniform -> SGPR neighbor base (saddr loads)
  if (t < 128) {
    int f = t;
    long sbase = (long)atom * 384 + f;
    float a0 = drA_g[sbase + 0];
    float a1 = drA_g[sbase + 128];
    float a2 = drA_g[sbase + 256];
    for (int m = 0; m < 64; ++m) {
      int nbm = __builtin_amdgcn_readfirstlane(nbl[m]);
      float ev = maskl[m] * (float)msg[m * LDA_T + f];
      const float* p = drA_g + bbase3 + (long)nbm * 384 + f;
      a0 += ev * p[0];
      a1 += ev * p[128];
      a2 += ev * p[256];
    }
    drB_g[sbase + 0] = a0;
    drB_g[sbase + 128] = a1;
    drB_g[sbase + 256] = a2;
  }
}

// ---------- finalize: dr += esu*upd_f ; s -= isu*(f_eq.dr) ; layernorm ----------
__global__ __launch_bounds__(128) void nn_finalize_kernel(float* __restrict__ s_g,
    const float* __restrict__ esu, const float* __restrict__ isu,
    const float* __restrict__ updf, const float* __restrict__ drB,
    float* __restrict__ drA, const float* __restrict__ feq,
    const float* __restrict__ lng, const float* __restrict__ lnb) {
  int f = threadIdx.x;
  long atom = blockIdx.x;
  long b3 = atom * 384, b1 = atom * 128;
  float e = esu[b1 + f], is = isu[b1 + f];
  float dot = 0.0f;
#pragma unroll
  for (int d = 0; d < 3; ++d) {
    long o = b3 + d * 128 + f;
    float dn = drB[o] + e * updf[o];
    drA[o] = dn;
    dot += feq[o] * dn;
  }
  float s2 = s_g[b1 + f] - is * dot;
  __shared__ float red[128];
  red[f] = s2; __syncthreads();
  for (int off = 64; off > 0; off >>= 1) { if (f < off) red[f] += red[f + off]; __syncthreads(); }
  float mean = red[0] / 128.0f; __syncthreads();
  float dv = s2 - mean;
  red[f] = dv * dv; __syncthreads();
  for (int off = 64; off > 0; off >>= 1) { if (f < off) red[f] += red[f + off]; __syncthreads(); }
  float var = red[0] / 128.0f;
  s_g[b1 + f] = dv * rsqrtf(var + 1e-5f) * lng[f] + lnb[f];
}

// ---------- head ----------
__global__ __launch_bounds__(256) void nn_head_kernel(const float* __restrict__ s_g,
    const _Float16* __restrict__ W1t, const float* __restrict__ B1,
    const _Float16* __restrict__ W2t, const float* __restrict__ B2,
    const float* __restrict__ W3, const float* __restrict__ B3,
    const float* __restrict__ nsc, const float* __restrict__ nsh,
    const int* __restrict__ an, float* __restrict__ out) {
  __shared__ __align__(16) _Float16 Xh[64 * LDA_T];
  __shared__ __align__(16) _Float16 wgt[128 * LDW_T];
  __shared__ __align__(16) _Float16 hid[64 * LDA_T];
  __shared__ __align__(16) _Float16 hid2[64 * 64];
  __shared__ float rowval[64];
  int t = threadIdx.x, w = t >> 5, lane = t & 31;
  int r0 = (w >> 1) * 16;
  int khi = (lane >> 4) & 1, cl = lane & 15;
  long row0 = (long)blockIdx.x * 64;
  stage_rows_f16(s_g, row0, Xh, t);
  v8f acc[4];
  gemm_tiles<128>(Xh, LDA_T, W1t, 128, wgt, acc, t, w, lane);
  {
    int cbase = (w & 1) * 64;
#pragma unroll
    for (int j = 0; j < 4; ++j) {
      int col = cbase + j * 16 + cl;
      float bb = B1[col];
#pragma unroll
      for (int r = 0; r < 8; ++r)
        hid[(r0 + r + khi * 8) * LDA_T + col] = (_Float16)silu_f(acc[j][r] + bb);
    }
  }
  gemm_tiles<64>(hid, LDA_T, W2t, 128, wgt, acc, t, w, lane);
  {
    int cbase = (w & 1) * 32;
#pragma unroll
    for (int j = 0; j < 2; ++j) {
      int col = cbase + j * 16 + cl;
      float bb = B2[col];
#pragma unroll
      for (int r = 0; r < 8; ++r)
        hid2[(r0 + r + khi * 8) * 64 + col] = (_Float16)silu_f(acc[j][r] + bb);
    }
  }
  __syncthreads();
  if (t < 64) {
    float a = B3[0];
    for (int j = 0; j < 64; ++j) a += (float)hid2[t * 64 + j] * W3[j];
    long ar = row0 + t;
    int z = an[ar];
    float v = nsc[z] * a + nsh[z];
    rowval[t] = (z > 0) ? v : 0.0f;
  }
  __syncthreads();
  if (t == 0) {
    float ssum = 0.0f;
    for (int r = 0; r < 64; ++r) ssum += rowval[r];
    atomicAdd(&out[(int)(row0 >> 9)], ssum);
  }
}

extern "C" void kernel_launch(void* const* d_in, const int* in_sizes, int n_in,
                              void* d_out, int out_size, void* d_ws, size_t ws_size,
                              hipStream_t stream) {
  (void)in_sizes; (void)n_in; (void)out_size; (void)ws_size;
  auto F = [&](int i) { return (const float*)d_in[i]; };
  const int* an    = (const int*)d_in[0];
  const int* nbr   = (const int*)d_in[1];
  const float* nms = F(2);
  const float* dst = F(3);
  const float* dvc = F(4);
  const float* emb = F(5);
  float* ws = (float*)d_ws;
  float* s    = ws;                 // 262144
  float* imn  = ws + 262144;
  float* esu  = ws + 524288;
  float* isu  = ws + 786432;
  float* feq  = ws + 1048576;       // 786432
  float* drA  = ws + 1835008;
  float* drB  = ws + 2621440;
  float* updf = ws + 3407872;       // ends 4194304 floats
  _Float16* wh = (_Float16*)(ws + 4194304);   // f16 weight arena (172032 halfs)
  float* out  = (float*)d_out;

  auto PACK = [&](const float* W, long off, int Ksrc, int Kpad, int N) {
    nn_pack_kernel<<<(Kpad * N + 255) / 256, 256, 0, stream>>>(W, wh + off, Ksrc, Kpad, N);
  };

  nn_init_kernel<<<3072, 256, 0, stream>>>(an, emb, s, feq, drA, out);
  for (int l = 0; l < 3; ++l) {
    int p = 6 + 23 * l;
    PACK(F(p + 2),  WOFF_IMN1, 128, 128, 128);
    PACK(F(p + 4),  WOFF_IMN2, 128, 128, 128);
    PACK(F(p + 11), WOFF_ESU1, 128, 128, 128);
    PACK(F(p + 13), WOFF_ESU2, 128, 128, 128);
    PACK(F(p + 17), WOFF_ISU1, 128, 128, 128);
    PACK(F(p + 19), WOFF_ISU2, 128, 128, 128);
    PACK(F(p + 7),  WOFF_EMF1, 128, 128, 128);
    PACK(F(p + 9),  WOFF_EMF2, 128, 128, 128);
    PACK(F(p + 15), WOFF_EME1, 128, 128, 128);
    PACK(F(p + 16), WOFF_EME2, 128, 128, 128);
    PACK(F(p + 0),  WOFF_IME,  20,  64,  128);
    nn_mlp2_kernel<<<32, 256, 0, stream>>>(s, wh + WOFF_IMN1, F(p + 3), wh + WOFF_IMN2, F(p + 5), imn);
    nn_mega_kernel<<<2048, 256, 0, stream>>>(dst, dvc, nms, nbr, imn, s, feq, drA, drB, updf,
        wh + WOFF_IME, F(p + 1), F(p + 6),
        wh + WOFF_EMF1, F(p + 8), wh + WOFF_EMF2, F(p + 10),
        wh + WOFF_EME1, wh + WOFF_EME2);
    nn_mlp2_kernel<<<32, 256, 0, stream>>>(s, wh + WOFF_ESU1, F(p + 12), wh + WOFF_ESU2, F(p + 14), esu);
    nn_mlp2_kernel<<<32, 256, 0, stream>>>(s, wh + WOFF_ISU1, F(p + 18), wh + WOFF_ISU2, F(p + 20), isu);
    nn_finalize_kernel<<<2048, 128, 0, stream>>>(s, esu, isu, updf, drB, drA, feq,
        F(p + 21), F(p + 22));
  }
  PACK(F(75), 0, 128, 128, 128);       // head_w1
  PACK(F(77), 16384, 128, 128, 64);    // head_w2
  nn_head_kernel<<<32, 256, 0, stream>>>(s, wh, F(76), wh + 16384, F(78), F(79), F(80),
                                         F(81), F(82), an, out);
}
